// RiemannCurvatureExtractor_51273319580338
// MI455X (gfx1250) — compile-verified, matrix operand order checked by reading, not run on hardware
//
#include <hip/hip_runtime.h>

// ---------------- types for WMMA ----------------
typedef __attribute__((ext_vector_type(16))) __bf16         v16bf;
typedef __attribute__((ext_vector_type(8)))  float          v8f;
typedef __attribute__((ext_vector_type(8)))  unsigned short u16x8;
typedef __attribute__((ext_vector_type(16))) unsigned short u16x16;

// problem dims (fixed by reference)
#define B_   8
#define C_   64
#define H_   256
#define W_   256
#define HW_  65536
#define NELEM (B_ * C_ * HW_)         // 33554432
#define NKB  18                       // 576 / 32 K-blocks (tap-major: k = tap*64 + ci)
#define WPK_ELEMS (4 * NKB * 32 * 16) // 36864 packed bf16 weights

// conv tile config: block computes 64 (C_out) x 128 (w) outputs for one (b,h)
#define WTILE 128
#define LDSW  (WTILE + 2)             // 130 staged columns (halo)

__device__ __forceinline__ unsigned short f2bf(float f) {
    union { float f; unsigned int u; } v; v.f = f;
    unsigned int u = v.u;
    return (unsigned short)((u + 0x7FFFu + ((u >> 16) & 1u)) >> 16); // RNE
}

// K interleave inside a 32-wide K block for 16-bit A/B fragments (ISA 7.12.2):
// lane-half 0 holds K {0..7, 16..23}, lane-half 1 holds K {8..15, 24..31}.
__device__ __forceinline__ int koff(int j, int half) {
    return (j < 8) ? (half * 8 + j) : (8 + half * 8 + j);
}

// ---------------- pack conv weights into A-fragment order ----------------
// Wpk[((mt*18 + kb)*32 + lane)*16 + j] = bf16(conv_w[co][ci][kh][kw])
//   co = mt*16 + (lane&15), kb -> tap t = kb>>1, c0 = (kb&1)*32, ci = c0 + koff(j, lane>>4)
__global__ void pack_weights(const float* __restrict__ w, unsigned short* __restrict__ wpk) {
    int idx = blockIdx.x * 256 + threadIdx.x;
    if (idx >= WPK_ELEMS) return;
    int j    = idx & 15;
    int lane = (idx >> 4) & 31;
    int kb   = (idx >> 9) % NKB;
    int mt   = (idx >> 9) / NKB;
    int half = lane >> 4, m = lane & 15;
    int t  = kb >> 1;
    int c0 = (kb & 1) * 32;
    int ci = c0 + koff(j, half);
    int co = mt * 16 + m;
    int kh = t / 3, kw = t - kh * 3;
    wpk[idx] = f2bf(w[((co * 64 + ci) * 3 + kh) * 3 + kw]);
}

// ---------------- implicit-GEMM conv3x3 via bf16 WMMA ----------------
// block: (w-tile of 128, h, b); 8 waves = 4 M-tiles x 2 N-groups; 4 accumulators/wave.
__global__ __launch_bounds__(256)
void conv_wmma(const float* __restrict__ x, const unsigned short* __restrict__ wpk,
               float* __restrict__ y) {
    __shared__ unsigned short lds[3 * LDSW * 64]; // patch [dh][w(130)][ci(64)] bf16, 49920 B
    const int tid = threadIdx.x;
    const int w0  = blockIdx.x * WTILE;
    const int h   = blockIdx.y;
    const int b   = blockIdx.z;

    // stage zero-padded 3 x 130 x 64 patch; coalesced along w per (ci,dh)
    for (int idx = tid; idx < 3 * LDSW * 64; idx += 256) {
        int ci  = idx / (3 * LDSW);
        int rem = idx - ci * (3 * LDSW);
        int dh  = rem / LDSW;
        int w   = rem - dh * LDSW;
        int hg  = h - 1 + dh;
        int wg  = w0 - 1 + w;
        float val = 0.0f;
        if ((unsigned)hg < 256u && (unsigned)wg < 256u)
            val = x[((b * 64 + ci) << 16) + (hg << 8) + wg];
        lds[(dh * LDSW + w) * 64 + ci] = f2bf(val);
    }
    __syncthreads();

    const int wave = tid >> 5;
    const int lane = tid & 31;
    const int half = lane >> 4;
    const int nn   = lane & 15;
    const int mt   = wave >> 1;        // M tile 0..3 (co = mt*16 + ..)
    const int ng   = wave & 1;         // N group: columns ng*64 .. ng*64+63

    v8f acc[4] = {v8f{}, v8f{}, v8f{}, v8f{}};

    for (int kb = 0; kb < NKB; ++kb) {
        int t  = kb >> 1;
        int dh = t / 3, dw = t - dh * 3;
        int c0 = (kb & 1) * 32;

        // A fragment: 16 contiguous bf16 per lane from packed weights (shared by 4 WMMAs)
        const unsigned short* aptr = wpk + (((mt * NKB + kb) * 32 + lane) << 4);
        u16x16 araw;
        *((u16x8*)&araw)       = ((const u16x8*)aptr)[0];
        *(((u16x8*)&araw) + 1) = ((const u16x8*)aptr)[1];
        v16bf afrag = __builtin_bit_cast(v16bf, araw);

#pragma unroll
        for (int s = 0; s < 4; ++s) {
            int widx = ng * 64 + s * 16 + nn + dw;
            const unsigned short* bptr = &lds[(dh * LDSW + widx) * 64 + c0 + half * 8];
            u16x16 braw;
            *((u16x8*)&braw)       = *(const u16x8*)bptr;        // K = half*8 + {0..7}
            *(((u16x8*)&braw) + 1) = *(const u16x8*)(bptr + 16); // K = 16 + half*8 + {0..7}
            v16bf bfrag = __builtin_bit_cast(v16bf, braw);
            acc[s] = __builtin_amdgcn_wmma_f32_16x16x32_bf16(
                false, afrag, false, bfrag, (short)0, acc[s], false, false);
        }
    }

    // C/D layout: VGPR r -> M = half*8 + r, N = lane&15
#pragma unroll
    for (int s = 0; s < 4; ++s) {
        int ncol = w0 + ng * 64 + s * 16 + nn;
#pragma unroll
        for (int r = 0; r < 8; ++r) {
            int co  = mt * 16 + half * 8 + r;
            int row = ((b * 64 + co) << 16) + (h << 8);
            y[row + ncol] = acc[s][r];
        }
    }
}

// ---------------- BN statistics (deterministic tree reduction) ----------------
__global__ __launch_bounds__(256)
void bn_stats(const float* __restrict__ y, float* __restrict__ stats) {
    __shared__ float s_s[256], s_q[256];
    int c = blockIdx.x, tid = threadIdx.x;
    float s = 0.f, q = 0.f;
    for (int b = 0; b < B_; ++b) {
        const float* p = y + (size_t)(b * 64 + c) * HW_;
        for (int i = tid * 4; i < HW_; i += 1024) {
            float4 v = *(const float4*)(p + i);
            s += v.x + v.y + v.z + v.w;
            q += v.x * v.x + v.y * v.y + v.z * v.z + v.w * v.w;
        }
    }
    s_s[tid] = s; s_q[tid] = q; __syncthreads();
    for (int off = 128; off > 0; off >>= 1) {
        if (tid < off) { s_s[tid] += s_s[tid + off]; s_q[tid] += s_q[tid + off]; }
        __syncthreads();
    }
    if (tid == 0) {
        const float inv = 1.0f / 524288.0f;
        float mean = s_s[0] * inv;
        float var  = s_q[0] * inv - mean * mean;
        stats[c]      = mean;
        stats[64 + c] = rsqrtf(var + 1e-5f);
    }
}

// ---------------- BN + ReLU, float4 (conv bias cancels in BN) ----------------
__global__ void bn_relu(const float4* __restrict__ y, const float* __restrict__ stats,
                        const float* __restrict__ gamma, const float* __restrict__ beta,
                        float4* __restrict__ out) {
    int i4 = blockIdx.x * 256 + threadIdx.x;           // NELEM/4 groups
    int c = (i4 >> 14) & 63;                           // 16384 float4 per channel-plane
    float mu = stats[c], is = stats[64 + c] * gamma[c], bb = beta[c];
    float4 v = y[i4];
    float4 r;
    r.x = (v.x - mu) * is + bb; r.x = r.x > 0.f ? r.x : 0.f;
    r.y = (v.y - mu) * is + bb; r.y = r.y > 0.f ? r.y : 0.f;
    r.z = (v.z - mu) * is + bb; r.z = r.z > 0.f ? r.z : 0.f;
    r.w = (v.w - mu) * is + bb; r.w = r.w > 0.f ? r.w : 0.f;
    out[i4] = r;
}

// ---------------- circular 3x3 convs (== the reference's FFT products) ----------------
// Each thread produces 4 consecutive w outputs; 6-wide row windows shared by both filters.
__global__ void grad3x3(const float* __restrict__ xr,
                        const float* __restrict__ fx, const float* __restrict__ fy,
                        float4* __restrict__ gx, float4* __restrict__ gy) {
    int i4 = blockIdx.x * 256 + threadIdx.x;           // NELEM/4 output groups
    int w4   = (i4 & 63) << 2;
    int h    = (i4 >> 6) & 255;
    int base = (i4 >> 14) << 16;
    float cfx[9], cfy[9];
#pragma unroll
    for (int k = 0; k < 9; ++k) { cfx[k] = fx[k]; cfy[k] = fy[k]; }
    float sx[4] = {0.f, 0.f, 0.f, 0.f}, sy[4] = {0.f, 0.f, 0.f, 0.f};
#pragma unroll
    for (int p = 0; p < 3; ++p) {
        int hh = ((h - p) & 255) << 8;
        float r[6];
#pragma unroll
        for (int i = 0; i < 6; ++i)
            r[i] = xr[base + hh + ((w4 - 2 + i) & 255)];
#pragma unroll
        for (int j = 0; j < 4; ++j)
#pragma unroll
            for (int q = 0; q < 3; ++q) {
                float v = r[j + 2 - q];
                sx[j] += cfx[p * 3 + q] * v;
                sy[j] += cfy[p * 3 + q] * v;
            }
    }
    gx[i4] = make_float4(sx[0], sx[1], sx[2], sx[3]);
    gy[i4] = make_float4(sy[0], sy[1], sy[2], sy[3]);
}

__global__ void curvature(const float* __restrict__ gx, const float* __restrict__ gy,
                          const float* __restrict__ fx, const float* __restrict__ fy,
                          float4* __restrict__ out) {
    int i4 = blockIdx.x * 256 + threadIdx.x;
    int w4   = (i4 & 63) << 2;
    int h    = (i4 >> 6) & 255;
    int base = (i4 >> 14) << 16;
    float cfx[9], cfy[9];
#pragma unroll
    for (int k = 0; k < 9; ++k) { cfx[k] = fx[k]; cfy[k] = fy[k]; }
    float sxx[4] = {0.f, 0.f, 0.f, 0.f}, syy[4] = {0.f, 0.f, 0.f, 0.f};
    float cgx[4], cgy[4];
#pragma unroll
    for (int p = 0; p < 3; ++p) {
        int hh = ((h - p) & 255) << 8;
        float rx[6], ry[6];
#pragma unroll
        for (int i = 0; i < 6; ++i) {
            int a = base + hh + ((w4 - 2 + i) & 255);
            rx[i] = gx[a];
            ry[i] = gy[a];
        }
        if (p == 0) {  // center row (q=0 offset): gx/gy at (h, w4+j) = r[j+2]
#pragma unroll
            for (int j = 0; j < 4; ++j) { cgx[j] = rx[j + 2]; cgy[j] = ry[j + 2]; }
        }
#pragma unroll
        for (int j = 0; j < 4; ++j)
#pragma unroll
            for (int q = 0; q < 3; ++q) {
                sxx[j] += cfx[p * 3 + q] * rx[j + 2 - q];
                syy[j] += cfy[p * 3 + q] * ry[j + 2 - q];
            }
    }
    float4 xv = out[i4];  // x_relu; each thread reads only its own elements
    float o[4];
    float xs[4] = {xv.x, xv.y, xv.z, xv.w};
#pragma unroll
    for (int j = 0; j < 4; ++j) {
        float a = cgx[j], c = cgy[j];
        o[j] = sxx[j] + syy[j] + a * a + c * c - 2.0f * a * c + xs[j];
    }
    out[i4] = make_float4(o[0], o[1], o[2], o[3]);
}

// ---------------- host launch ----------------
extern "C" void kernel_launch(void* const* d_in, const int* in_sizes, int n_in,
                              void* d_out, int out_size, void* d_ws, size_t ws_size,
                              hipStream_t stream) {
    const float* x      = (const float*)d_in[0];
    const float* conv_w = (const float*)d_in[1];
    // d_in[2] (conv_b) intentionally unused: constant bias cancels in BatchNorm
    const float* gamma  = (const float*)d_in[3];
    const float* beta   = (const float*)d_in[4];
    const float* filtx  = (const float*)d_in[5];
    const float* filty  = (const float*)d_in[6];
    float* out = (float*)d_out;

    // workspace layout
    float*          y     = (float*)d_ws;         // NELEM floats (conv out; reused as grad_x)
    float*          gybuf = y + NELEM;             // NELEM floats (grad_y)
    float*          stats = gybuf + NELEM;         // 128 floats (mean, invstd)
    unsigned short* wpk   = (unsigned short*)(stats + 128); // WPK_ELEMS bf16
    float*          gxbuf = y;                     // alias after bn_relu consumed y

    pack_weights<<<(WPK_ELEMS + 255) / 256, 256, 0, stream>>>(conv_w, wpk);
    conv_wmma<<<dim3(W_ / WTILE, H_, B_), 256, 0, stream>>>(x, wpk, y);
    bn_stats<<<64, 256, 0, stream>>>(y, stats);
    bn_relu<<<NELEM / 4 / 256, 256, 0, stream>>>((const float4*)y, stats, gamma, beta,
                                                 (float4*)out);
    grad3x3<<<NELEM / 4 / 256, 256, 0, stream>>>(out, filtx, filty,
                                                 (float4*)gxbuf, (float4*)gybuf);
    curvature<<<NELEM / 4 / 256, 256, 0, stream>>>(gxbuf, gybuf, filtx, filty,
                                                   (float4*)out);
}
